// ViT_35296041238901
// MI455X (gfx1250) — matrix operand order repaired and checked by reading, not compiled, not run on hardware
//
#include <hip/hip_runtime.h>
#include <hip/hip_bf16.h>
#include <math.h>

// ---------------------------------------------------------------------------
// ViT forward for MI455X (gfx1250, wave32). All GEMMs via v_wmma_f32_16x16x32_f16
// (f16 operands, f32 accumulate). Working set (~134MB) fits in the 192MB L2, so
// GEMM streams operands from L2 with 4x4 WMMA register blocking per wave
// (64x64/wave, 128x128/workgroup) -> 32 FLOP per fragment byte, and 16
// independent back-to-back WMMAs per K-step to hide WMMA RAW-hazard latency.
// Attention is fully WMMA with zero-padded K dims.
// ---------------------------------------------------------------------------

typedef __attribute__((ext_vector_type(16))) _Float16 v16h;
typedef __attribute__((ext_vector_type(8)))  float    v8f;

constexpr int kC   = 256;        // embed dim
constexpr int kT   = 65;         // tokens (64 patches + cls)
constexpr int kB   = 256;        // batch
constexpr int kM   = kB * kT;    // 16640 rows (= 130 * 128)
constexpr int kMpe = kB * 64;    // 16384 patch rows (= 128 * 128)
constexpr int kKpe = 768;        // 3*16*16 patch pixels
constexpr int kFF  = 1024;

union AFrag {
  v16h     v;
  uint4    q[2];
  _Float16 h[16];
};

__device__ inline v8f zero_v8f() {
  v8f z;
#pragma unroll
  for (int i = 0; i < 8; ++i) z[i] = 0.0f;
  return z;
}

// ---------------------------------------------------------------------------
// Generic WMMA GEMM:  C[M,N] = A[M,K] (f16, row-major) * Bt[N,K]^T (f16)
// EPI: 0 = store f32            1 = store f16 (+bias if given)
//      2 = store f16, bias+GELU 3 = store f32, bias + residual (f32)
// Block: 128 thr = 4 waves (2x2), wave -> 64x64 tile (4x4 WMMA tiles),
// workgroup -> 128x128 tile. M % 128 == 0, N % 128 == 0 guaranteed by caller.
// ---------------------------------------------------------------------------
template <int EPI>
__global__ __launch_bounds__(128) void gemm_wmma_kernel(
    const _Float16* __restrict__ A, const _Float16* __restrict__ Bt,
    float* __restrict__ outF32, _Float16* __restrict__ outF16,
    const float* __restrict__ bias, const float* __restrict__ resid,
    int M, int N, int K) {
  const int lane = threadIdx.x & 31;
  const int wave = threadIdx.x >> 5;
  const int half = lane >> 4;      // which 16-lane group
  const int lr   = lane & 15;
  const int rowBase = blockIdx.y * 128 + (wave >> 1) * 64;
  const int colBase = blockIdx.x * 128 + (wave & 1) * 64;

  // Per-lane base pointers for the 4 A row-tiles and 4 B col-tiles.
  // A 16x32 f16 layout: lane holds A[m][kb..kb+7] and A[m][kb+16..kb+23],
  //   kb = 8*half, m = lr                      -> two 16B loads per tile.
  // B 32x16 f16 layout: lane (col n=lr) holds B[kb..kb+7][n] and B[kb+16..kb+23][n],
  //   kb = 8*half, in Bt[n][k] row-major storage -> two 16B loads per tile.
  const _Float16* aRow[4];
  const _Float16* bRow[4];
#pragma unroll
  for (int i = 0; i < 4; ++i)
    aRow[i] = A + (size_t)(rowBase + i * 16 + lr) * K + half * 8;
#pragma unroll
  for (int j = 0; j < 4; ++j)
    bRow[j] = Bt + (size_t)(colBase + j * 16 + lr) * K + half * 8;

  v8f acc[4][4];
#pragma unroll
  for (int i = 0; i < 4; ++i)
#pragma unroll
    for (int j = 0; j < 4; ++j) acc[i][j] = zero_v8f();

  for (int k0 = 0; k0 < K; k0 += 32) {
    AFrag a[4], b[4];
#pragma unroll
    for (int i = 0; i < 4; ++i) {
      a[i].q[0] = *(const uint4*)(aRow[i] + k0);
      a[i].q[1] = *(const uint4*)(aRow[i] + k0 + 16);
    }
#pragma unroll
    for (int j = 0; j < 4; ++j) {
      b[j].q[0] = *(const uint4*)(bRow[j] + k0);
      b[j].q[1] = *(const uint4*)(bRow[j] + k0 + 16);
    }
    if (k0 + 32 < K) {
      // Prefetch next K-block of the streaming operand (activations).
#pragma unroll
      for (int i = 0; i < 4; ++i)
        __builtin_prefetch(aRow[i] + k0 + 32, 0, 1);
    }
#pragma unroll
    for (int i = 0; i < 4; ++i)
#pragma unroll
      for (int j = 0; j < 4; ++j)
        acc[i][j] = __builtin_amdgcn_wmma_f32_16x16x32_f16(
            false, a[i].v, false, b[j].v, (short)0, acc[i][j], false, false);
  }

  // Epilogue. D layout: VGPR e -> row (e + 8*half), col = lr.
#pragma unroll
  for (int j = 0; j < 4; ++j) {
    const int col = colBase + j * 16 + lr;
    const float bv = bias ? bias[col] : 0.0f;
#pragma unroll
    for (int i = 0; i < 4; ++i)
#pragma unroll
      for (int e = 0; e < 8; ++e) {
        const int row = rowBase + i * 16 + e + half * 8;
        float v = acc[i][j][e] + bv;
        if (EPI == 2) v = 0.5f * v * (1.0f + erff(v * 0.70710678118f));
        if (EPI == 3) v += resid[(size_t)row * N + col];
        if (EPI == 0 || EPI == 3) outF32[(size_t)row * N + col] = v;
        else                      outF16[(size_t)row * N + col] = (_Float16)v;
      }
  }
}

// ---------------------------------------------------------------------------
// Fused attention per (batch, head): one workgroup of 5 wave32 (one per
// 16-row query tile, T=65 padded to 80). Scores: K dim = HD=16 zero-padded to
// 32 (single WMMA). Softmax in D-register layout with width-16 shfl_xor
// reductions. P staged to LDS (f16, T padded to 96) to re-enter A layout,
// then O = P*V with 3 WMMA K-steps against LDS-staged V^T.
// qkv layout: [B, T, 768] f16 (q | k | v, each [H=16, HD=16] per token).
// o layout:   [B, T, 256] f16 (heads concatenated).
// ---------------------------------------------------------------------------
__global__ __launch_bounds__(160) void attn_kernel(
    const _Float16* __restrict__ qkv, _Float16* __restrict__ o) {
  __shared__ _Float16 Vt[16][96];      // Vt[d][s], s zero-padded past 65
  __shared__ _Float16 Pm[5][16][96];   // per-wave probability tile, K-padded
  const int b    = blockIdx.x >> 4;
  const int hh   = blockIdx.x & 15;
  const int tid  = threadIdx.x;
  const int wave = tid >> 5;
  const int lane = tid & 31;
  const int half = lane >> 4;
  const int lr   = lane & 15;
  const _Float16* base = qkv + (size_t)b * kT * 768 + hh * 16;

  for (int idx = tid; idx < 16 * 96; idx += 160) {
    const int d = idx / 96, s = idx % 96;
    Vt[d][s] = (s < kT) ? base[(size_t)s * 768 + 512 + d] : (_Float16)0.0f;
  }
  for (int idx = tid; idx < 5 * 16 * 16; idx += 160) {
    const int wv = idx >> 8, m = (idx >> 4) & 15, c2 = idx & 15;
    Pm[wv][m][80 + c2] = (_Float16)0.0f;   // zero K-padding columns 80..95
  }
  __syncthreads();

  // ---- S = Q K^T, HD padded 16 -> 32 ----
  AFrag qa;
#pragma unroll
  for (int i = 0; i < 16; ++i) qa.h[i] = (_Float16)0.0f;
  {
    const int t = wave * 16 + lr;
    if (half == 0 && t < kT) { qa.q[0] = *(const uint4*)(base + (size_t)t * 768); qa.q[1] = *(const uint4*)(base + (size_t)t * 768 + 8); }
  }
  v8f s[5];
#pragma unroll
  for (int j = 0; j < 5; ++j) {
    AFrag kb;
#pragma unroll
    for (int i = 0; i < 16; ++i) kb.h[i] = (_Float16)0.0f;
    const int si = j * 16 + lr;
    if (half == 0 && si < kT) {            // K rows 16..31 are zero padding
      const _Float16* kp = base + (size_t)si * 768 + 256;
      kb.q[0] = *(const uint4*)(kp);
      kb.q[1] = *(const uint4*)(kp + 8);
    }
    s[j] = __builtin_amdgcn_wmma_f32_16x16x32_f16(
        false, qa.v, false, kb.v, (short)0, zero_v8f(), false, false);
  }

  // ---- softmax over each row (16 lanes x 5 tiles hold one row) ----
  const float scale = 0.0625f;  // C^-0.5 = 256^-0.5 (source scales by n_embd)
#pragma unroll
  for (int e = 0; e < 8; ++e) {
    float vals[5];
    float mx = -1e30f;
#pragma unroll
    for (int j = 0; j < 5; ++j) {
      const int col = j * 16 + lr;
      const float vv = (col < kT) ? (float)s[j][e] * scale : -1e30f;
      vals[j] = vv;
      mx = fmaxf(mx, vv);
    }
#pragma unroll
    for (int off = 1; off < 16; off <<= 1)
      mx = fmaxf(mx, __shfl_xor(mx, off, 32));
    float sum = 0.0f;
#pragma unroll
    for (int j = 0; j < 5; ++j) {
      const float p = (vals[j] > -1e29f) ? __expf(vals[j] - mx) : 0.0f;
      vals[j] = p;
      sum += p;
    }
#pragma unroll
    for (int off = 1; off < 16; off <<= 1) sum += __shfl_xor(sum, off, 32);
    const float rinv = (sum > 0.0f) ? (1.0f / sum) : 0.0f;
    const int m = e + half * 8;
#pragma unroll
    for (int j = 0; j < 5; ++j)
      Pm[wave][m][j * 16 + lr] = (_Float16)(vals[j] * rinv);
  }
  __syncthreads();

  // ---- O = P V  (K = 96, 3 WMMA steps) ----
  v8f oacc = zero_v8f();
#pragma unroll
  for (int k0 = 0; k0 < 96; k0 += 32) {
    AFrag pa, vb;
    const _Float16* pp = &Pm[wave][lr][k0 + half * 8];
    pa.q[0] = *(const uint4*)(pp);
    pa.q[1] = *(const uint4*)(pp + 16);
    const _Float16* vp = &Vt[lr][k0 + half * 8];
    vb.q[0] = *(const uint4*)(vp);
    vb.q[1] = *(const uint4*)(vp + 16);
    oacc = __builtin_amdgcn_wmma_f32_16x16x32_f16(
        false, pa.v, false, vb.v, (short)0, oacc, false, false);
  }
#pragma unroll
  for (int e = 0; e < 8; ++e) {
    const int t = wave * 16 + e + half * 8;
    if (t < kT)
      o[((size_t)b * kT + t) * 256 + hh * 16 + lr] = (_Float16)oacc[e];
  }
}

// ---------------------------------------------------------------------------
// LayerNorm: one wave32 per 256-wide row (8 floats/lane, shfl_xor tree).
// ---------------------------------------------------------------------------
template <typename OutT>
__global__ __launch_bounds__(256) void ln_kernel(
    const float* __restrict__ x, const float* __restrict__ g,
    const float* __restrict__ bb, OutT* __restrict__ out,
    int nrows, long long strIn, long long strOut) {
  const int row = blockIdx.x * 8 + (threadIdx.x >> 5);
  if (row >= nrows) return;
  const int lane = threadIdx.x & 31;
  const float* xp = x + (size_t)row * strIn + lane * 8;
  float v[8];
  const float4 a0 = *(const float4*)(xp);
  const float4 a1 = *(const float4*)(xp + 4);
  v[0] = a0.x; v[1] = a0.y; v[2] = a0.z; v[3] = a0.w;
  v[4] = a1.x; v[5] = a1.y; v[6] = a1.z; v[7] = a1.w;
  float s = 0.0f;
#pragma unroll
  for (int i = 0; i < 8; ++i) s += v[i];
#pragma unroll
  for (int off = 1; off < 32; off <<= 1) s += __shfl_xor(s, off, 32);
  const float mu = s * (1.0f / 256.0f);
  float var = 0.0f;
#pragma unroll
  for (int i = 0; i < 8; ++i) { const float d = v[i] - mu; var += d * d; }
#pragma unroll
  for (int off = 1; off < 32; off <<= 1) var += __shfl_xor(var, off, 32);
  const float rs = rsqrtf(var * (1.0f / 256.0f) + 1e-5f);
#pragma unroll
  for (int i = 0; i < 8; ++i) {
    const int c = lane * 8 + i;
    out[(size_t)row * strOut + c] = (OutT)((v[i] - mu) * rs * g[c] + bb[c]);
  }
}

// ---------------------------------------------------------------------------
// Helpers: weight f32->f16, im2col for the 16x16/stride-16 conv, token build.
// ---------------------------------------------------------------------------
__global__ void f32_to_f16_kernel(const float* __restrict__ in,
                                  _Float16* __restrict__ out, size_t n) {
  const size_t i = (size_t)blockIdx.x * 256 + threadIdx.x;
  if (i < n) out[i] = (_Float16)in[i];
}

__global__ void im2col_kernel(const float* __restrict__ X,
                              _Float16* __restrict__ A) {
  const size_t idx = (size_t)blockIdx.x * 256 + threadIdx.x;
  if (idx >= (size_t)kMpe * kKpe) return;
  const int m = (int)(idx / kKpe), k = (int)(idx % kKpe);
  const int b = m >> 6, p = m & 63, py = p >> 3, px = p & 7;
  const int c = k >> 8, r = k & 255, iy = r >> 4, ix = r & 15;
  A[idx] = (_Float16)X[(((size_t)b * 3 + c) * 128 + py * 16 + iy) * 128 +
                       px * 16 + ix];
}

__global__ void assemble_kernel(const float* __restrict__ pe,
                                const float* __restrict__ convb,
                                const float* __restrict__ cls,
                                const float* __restrict__ pos,
                                float* __restrict__ x) {
  const size_t idx = (size_t)blockIdx.x * 256 + threadIdx.x;
  if (idx >= (size_t)kM * kC) return;
  const int c = (int)(idx & 255);
  const size_t row = idx >> 8;
  const int t = (int)(row % kT), b = (int)(row / kT);
  float v;
  if (t == 0) v = cls[c] + pos[c];
  else v = pe[((size_t)b * 64 + (t - 1)) * 256 + c] + convb[c] +
           pos[(size_t)t * 256 + c];
  x[idx] = v;
}

// ---------------------------------------------------------------------------
extern "C" void kernel_launch(void* const* d_in, const int* in_sizes, int n_in,
                              void* d_out, int out_size, void* d_ws,
                              size_t ws_size, hipStream_t stream) {
  (void)in_sizes; (void)n_in; (void)out_size; (void)ws_size;
  const float* X      = (const float*)d_in[0];
  const float* conv_w = (const float*)d_in[1];
  const float* conv_b = (const float*)d_in[2];
  const float* cls    = (const float*)d_in[3];
  const float* pos    = (const float*)d_in[4];
  const float* lnf_g  = (const float*)d_in[5];
  const float* lnf_b  = (const float*)d_in[6];
  auto BP = [&](int blk, int j) { return (const float*)d_in[7 + blk * 13 + j]; };
  // block param order: 0 ln1_g, 1 ln1_b, 2 wq, 3 wk, 4 wv, 5 proj_w,
  //                    6 proj_b, 7 ln2_g, 8 ln2_b, 9 w1, 10 b1, 11 w2, 12 b2

  char* ws = (char*)d_ws;
  size_t off = 0;
  auto alloc = [&](size_t bytes) {
    void* p = ws + off;
    off += (bytes + 255) & ~(size_t)255;
    return p;
  };

  _Float16* wc = (_Float16*)alloc((size_t)256 * 768 * 2);
  _Float16 *wqkv[4], *wpr[4], *w1[4], *w2[4];
  for (int i = 0; i < 4; ++i) {
    wqkv[i] = (_Float16*)alloc((size_t)768 * 256 * 2);
    wpr[i]  = (_Float16*)alloc((size_t)256 * 256 * 2);
    w1[i]   = (_Float16*)alloc((size_t)1024 * 256 * 2);
    w2[i]   = (_Float16*)alloc((size_t)256 * 1024 * 2);
  }
  _Float16* Ape = (_Float16*)alloc((size_t)kMpe * 768 * 2);
  float*    pe  = (float*)alloc((size_t)kMpe * 256 * 4);
  float*    x   = (float*)alloc((size_t)kM * 256 * 4);
  _Float16* h   = (_Float16*)alloc((size_t)kM * 256 * 2);  // LN out / attn out
  _Float16* qkv = (_Float16*)alloc((size_t)kM * 768 * 2);
  _Float16* ff  = (_Float16*)alloc((size_t)kM * 1024 * 2);

  auto cvt = [&](const float* src, _Float16* dst, size_t n) {
    f32_to_f16_kernel<<<(unsigned)((n + 255) / 256), 256, 0, stream>>>(src, dst, n);
  };

  // ---- weights to f16 (kept [out,in] row-major: B-operand contiguous) ----
  cvt(conv_w, wc, (size_t)256 * 768);
  for (int i = 0; i < 4; ++i) {
    cvt(BP(i, 2),  wqkv[i],          (size_t)256 * 256);  // wq
    cvt(BP(i, 3),  wqkv[i] + 65536,  (size_t)256 * 256);  // wk
    cvt(BP(i, 4),  wqkv[i] + 131072, (size_t)256 * 256);  // wv
    cvt(BP(i, 5),  wpr[i],           (size_t)256 * 256);
    cvt(BP(i, 9),  w1[i],            (size_t)1024 * 256);
    cvt(BP(i, 11), w2[i],            (size_t)256 * 1024);
  }

  // ---- patch embed as GEMM [16384,768]x[768,256] ----
  im2col_kernel<<<(kMpe * kKpe) / 256, 256, 0, stream>>>(X, Ape);
  gemm_wmma_kernel<0><<<dim3(256 / 128, kMpe / 128), 128, 0, stream>>>(
      Ape, wc, pe, nullptr, nullptr, nullptr, kMpe, 256, 768);
  assemble_kernel<<<(kM * kC) / 256, 256, 0, stream>>>(pe, conv_b, cls, pos, x);

  // ---- transformer blocks ----
  for (int i = 0; i < 4; ++i) {
    ln_kernel<_Float16><<<kM / 8, 256, 0, stream>>>(x, BP(i, 0), BP(i, 1), h,
                                                    kM, 256, 256);
    gemm_wmma_kernel<1><<<dim3(768 / 128, kM / 128), 128, 0, stream>>>(
        h, wqkv[i], nullptr, qkv, nullptr, nullptr, kM, 768, 256);
    attn_kernel<<<kB * 16, 160, 0, stream>>>(qkv, h);  // h reused as O
    gemm_wmma_kernel<3><<<dim3(256 / 128, kM / 128), 128, 0, stream>>>(
        h, wpr[i], x, nullptr, BP(i, 6), x, kM, 256, 256);
    ln_kernel<_Float16><<<kM / 8, 256, 0, stream>>>(x, BP(i, 7), BP(i, 8), h,
                                                    kM, 256, 256);
    gemm_wmma_kernel<2><<<dim3(1024 / 128, kM / 128), 128, 0, stream>>>(
        h, w1[i], nullptr, ff, BP(i, 10), nullptr, kM, 1024, 256);
    gemm_wmma_kernel<3><<<dim3(256 / 128, kM / 128), 128, 0, stream>>>(
        ff, w2[i], x, nullptr, BP(i, 12), x, kM, 256, 1024);
  }

  // ---- final LN on cls token rows -> d_out [256,256] f32 ----
  ln_kernel<float><<<kB / 8, 256, 0, stream>>>(
      x, lnf_g, lnf_b, (float*)d_out, kB, (long long)kT * 256, 256);
}
